// GTM_attention_71365176590894
// MI455X (gfx1250) — matrix operand order repaired and checked by reading, not compiled, not run on hardware
//
#include <hip/hip_runtime.h>

// ---------------------------------------------------------------------------
// GAT-style attention for MI455X (gfx1250, wave32, WMMA).
//
// Shapes: B=8, N=2048, DIN=DOUT=128, DA=2, neg_slope=0.2.
//
// Softmax shift invariance: softmax_j(lr_row[i] + lr_col[j]) ==
// softmax_j(lr_col[j]) (masked entries are -1e9 and underflow to exactly 0
// in f32 either way). So lr_row / Wr are dead and
//   atten[b,i,j] = mask * E[b,j] / S[b,i],  E[b,j] = exp(lr_col[b,j] - max_b).
// The N^2 logits tensor is never materialized; the only N^2 memory traffic
// is one streaming pass over mask (134 MB -> ~6 us at 23.3 TB/s), and the
// 8.6 GFLOP atten@xv matmul runs on v_wmma_f32_16x16x32_bf16.
// ---------------------------------------------------------------------------

typedef __attribute__((ext_vector_type(16))) __bf16  v16bf;
typedef __attribute__((ext_vector_type(8)))  float   v8f;

union BFu {
    v16bf    v;
    uint4    q[2];
    unsigned d[8];
};

__device__ __forceinline__ unsigned short f2bf(float f) {
    unsigned u = __float_as_uint(f);
    unsigned r = u + 0x7FFFu + ((u >> 16) & 1u);   // round-to-nearest-even
    return (unsigned short)(r >> 16);
}

// select bf16 pair by two mask ints (mask>0 keeps, else 0)
__device__ __forceinline__ unsigned sel2(int ma, int mb, unsigned epair) {
    unsigned lo = (ma > 0) ? (epair & 0x0000FFFFu) : 0u;
    unsigned hi = (mb > 0) ? (epair & 0xFFFF0000u) : 0u;
    return lo | hi;
}

// ---------------------------------------------------------------------------
// K1: lr_col[b,j] = sum_a leaky_relu(dot(x[b,j,:], Wc[a,:])) * Wcat[DA+a]
// ---------------------------------------------------------------------------
__global__ __launch_bounds__(256) void k_lrcol(const float* __restrict__ x,
                                               const float* __restrict__ Wc,
                                               const float* __restrict__ Wcat,
                                               float* __restrict__ lrcol) {
    int gid = blockIdx.x * 256 + threadIdx.x;          // 0..16383
    const float4* xr = (const float4*)(x + (size_t)gid * 128);
    const float4* w0 = (const float4*)(Wc);
    const float4* w1 = (const float4*)(Wc + 128);
    float a0 = 0.f, a1 = 0.f;
    for (int d = 0; d < 32; ++d) {
        float4 xv = xr[d];
        float4 c0 = w0[d];
        float4 c1 = w1[d];
        a0 += xv.x * c0.x + xv.y * c0.y + xv.z * c0.z + xv.w * c0.w;
        a1 += xv.x * c1.x + xv.y * c1.y + xv.z * c1.z + xv.w * c1.w;
    }
    float l0 = (a0 > 0.f) ? a0 : 0.2f * a0;
    float l1 = (a1 > 0.f) ? a1 : 0.2f * a1;
    lrcol[gid] = l0 * Wcat[2] + l1 * Wcat[3];
}

// ---------------------------------------------------------------------------
// K2: per-batch max reduce, then Ebf[b,j] = bf16(exp(lr_col - max_b)).
// ---------------------------------------------------------------------------
__global__ __launch_bounds__(256) void k_expnorm(const float* __restrict__ lrcol,
                                                 unsigned short* __restrict__ Ebf) {
    __shared__ float red[256];
    const int b = blockIdx.x, tid = threadIdx.x;
    const float* src = lrcol + b * 2048;
    float v[8];
    float mx = -3.4e38f;
#pragma unroll
    for (int i = 0; i < 8; ++i) {
        v[i] = src[tid + i * 256];
        mx = fmaxf(mx, v[i]);
    }
    red[tid] = mx;
    __syncthreads();
    for (int s = 128; s > 0; s >>= 1) {
        if (tid < s) red[tid] = fmaxf(red[tid], red[tid + s]);
        __syncthreads();
    }
    mx = red[0];
    unsigned short* dst = Ebf + b * 2048;
#pragma unroll
    for (int i = 0; i < 8; ++i) dst[tid + i * 256] = f2bf(__expf(v[i] - mx));
}

// ---------------------------------------------------------------------------
// K3: xv[b,j,o] = dot(x[b,j,:], Wx[o,:]) + bx[o], stored bf16 in WMMA
// B-fragment swizzled order:
//   idx = (((b*64 + j/32)*8 + o/16)*512) + lane*16 + e
//   lane = 16*((j>>3)&1) + (o&15),  e = (j&7) + 8*((j>>4)&1)
// so stage-4 B-operand loads are two coalesced b128 per lane per fragment.
// ---------------------------------------------------------------------------
__global__ __launch_bounds__(256) void k_xvswiz(const float* __restrict__ x,
                                                const float* __restrict__ Wx,
                                                const float* __restrict__ bx,
                                                unsigned short* __restrict__ swiz) {
    __shared__ float xs[32 * 128];
    const int tid = threadIdx.x;
    const int g0  = blockIdx.x * 32;       // global row base
    const int b   = g0 >> 11;
    const int jb  = g0 & 2047;             // row base within batch

    const float4* xsrc = (const float4*)(x + (size_t)g0 * 128);
    float4* xd = (float4*)xs;
#pragma unroll
    for (int k = 0; k < 4; ++k) xd[tid + k * 256] = xsrc[tid + k * 256];
    __syncthreads();

    const int o    = tid & 127;
    const int jset = tid >> 7;             // 0 or 1 -> rows jset*16..+15
    float acc[16];
#pragma unroll
    for (int r = 0; r < 16; ++r) acc[r] = 0.f;

    const float4* wrow  = (const float4*)(Wx + (size_t)o * 128);
    const float4* xrow0 = (const float4*)(xs + (jset * 16) * 128);
    for (int d4 = 0; d4 < 32; ++d4) {
        float4 w = wrow[d4];
#pragma unroll
        for (int r = 0; r < 16; ++r) {
            float4 xv = xrow0[r * 32 + d4];
            acc[r] += w.x * xv.x + w.y * xv.y + w.z * xv.z + w.w * xv.w;
        }
    }
    const float bo = bx[o];
    const int ct = o >> 4;
#pragma unroll
    for (int r = 0; r < 16; ++r) {
        int j = jb + jset * 16 + r;
        int kt = j >> 5;
        int e = (j & 7) + (((j >> 4) & 1) << 3);
        int lane = (((j >> 3) & 1) << 4) + (o & 15);
        size_t idx = ((((size_t)b * 64 + kt) * 8 + ct) * 512) + lane * 16 + e;
        swiz[idx] = f2bf(acc[r] + bo);
    }
}

// ---------------------------------------------------------------------------
// K4: out[b,i,:] = (sum_j mask*E[j]*xv[j,:]) / (sum_j mask*E[j])
// 1024 blocks (one 16-row i-tile of one batch), 4 waves split the j range.
// A-fragment (16x32 bf16, ISA layout: k = (e<8?e:e+8) + 8*(lane>=16)) from
// 4 b128 mask loads + 2 b128 E loads per k-step.  B-fragments rotate through
// 4 register buffers so loads run ~4 WMMAs ahead of consumption; the
// ones-column WMMA (softmax denominator S) issues first (no load dep).
// Mask lines for step+2 are software-prefetched (pure HBM stream; one 128 B
// line covers all 32 k's of a row, lanes cover the 16 rows).
// ---------------------------------------------------------------------------
__global__ __launch_bounds__(128) void k_attn(const int* __restrict__ mask,
                                              const unsigned short* __restrict__ Ebf,
                                              const unsigned short* __restrict__ swiz,
                                              float* __restrict__ out) {
    __shared__ float part[4][16][128];
    __shared__ float pS[4][16];

    const int b    = blockIdx.x >> 7;
    const int i0   = (blockIdx.x & 127) << 4;
    const int tid  = threadIdx.x;
    const int wave = tid >> 5;
    const int lane = tid & 31;
    const int lhalf = lane >> 4;           // 0/1
    const int l15   = lane & 15;
    const int m     = l15;                 // A-matrix row owned by this lane
    const int koff0 = lhalf ? 8 : 0;       // A layout K offset for e=0..7

    v8f zf = {0.f, 0.f, 0.f, 0.f, 0.f, 0.f, 0.f, 0.f};
    v8f acc[9];
#pragma unroll
    for (int t = 0; t < 9; ++t) acc[t] = zf;

    // constant ones-column B fragment: column n=128+(l15) -> 1.0 iff l15==0
    BFu b1;
    {
        unsigned ones = (l15 == 0) ? 0x3F803F80u : 0u;
#pragma unroll
        for (int w = 0; w < 8; ++w) b1.d[w] = ones;
    }

    const int* mrow = mask + ((size_t)b * 2048 + (i0 + m)) * 2048;
    const unsigned short* Erow  = Ebf + b * 2048;
    const unsigned short* sbase = swiz + (size_t)b * 64 * 8 * 512 + (size_t)lane * 16;

    for (int step = 0; step < 16; ++step) {
        const int kt = wave * 16 + step;   // 32-wide k-tile index
        const int j0 = kt << 5;

        // prefetch the mask lines this wave needs 2 steps from now
        __builtin_prefetch(mrow + j0 + 64 + koff0, 0, 0);

        // ---- A-tile loads (mask + E) ----
        const int* mp = mrow + j0 + koff0;
        uint4 m0 = *(const uint4*)(mp);
        uint4 m1 = *(const uint4*)(mp + 4);
        uint4 m2 = *(const uint4*)(mp + 16);
        uint4 m3 = *(const uint4*)(mp + 20);
        uint4 e0 = *(const uint4*)(Erow + j0 + koff0);        // e = 0..7
        uint4 e1 = *(const uint4*)(Erow + j0 + koff0 + 16);   // e = 8..15

        // ---- preload first 4 B fragments (rotating buffers) ----
        const unsigned short* bkt = sbase + (size_t)kt * (8 * 512);
        BFu bfr[4];
#pragma unroll
        for (int p = 0; p < 4; ++p) {
            const uint4* q = (const uint4*)(bkt + p * 512);
            bfr[p].q[0] = q[0];
            bfr[p].q[1] = q[1];
        }

        // ---- build A fragment (VALU overlaps outstanding loads) ----
        BFu af;
        af.d[0] = sel2((int)m0.x, (int)m0.y, e0.x);
        af.d[1] = sel2((int)m0.z, (int)m0.w, e0.y);
        af.d[2] = sel2((int)m1.x, (int)m1.y, e0.z);
        af.d[3] = sel2((int)m1.z, (int)m1.w, e0.w);
        af.d[4] = sel2((int)m2.x, (int)m2.y, e1.x);
        af.d[5] = sel2((int)m2.z, (int)m2.w, e1.y);
        af.d[6] = sel2((int)m3.x, (int)m3.y, e1.z);
        af.d[7] = sel2((int)m3.z, (int)m3.w, e1.w);

        // ---- ones-column WMMA first: no load dependency ----
        acc[8] = __builtin_amdgcn_wmma_f32_16x16x32_bf16(
            false, af.v, false, b1.v, (short)0, acc[8], false, false);

        // ---- 8 data WMMAs, loads running 4 fragments ahead ----
#pragma unroll
        for (int ct = 0; ct < 8; ++ct) {
            v16bf bm = bfr[ct & 3].v;
            if (ct < 4) {
                const uint4* q = (const uint4*)(bkt + (ct + 4) * 512);
                bfr[ct & 3].q[0] = q[0];
                bfr[ct & 3].q[1] = q[1];
            }
            acc[ct] = __builtin_amdgcn_wmma_f32_16x16x32_bf16(
                false, af.v, false, bm, (short)0, acc[ct], false, false);
        }
    }

    // ---- cross-wave reduction via LDS ----
#pragma unroll
    for (int r = 0; r < 8; ++r) {
        int mm = r + (lhalf << 3);         // C/D layout: m = r + 8*(lane>=16)
#pragma unroll
        for (int ct = 0; ct < 8; ++ct)
            part[wave][mm][ct * 16 + l15] = acc[ct][r];
        if (l15 == 0) pS[wave][mm] = acc[8][r];
    }
    __syncthreads();

    float* orow = out + ((size_t)b * 2048 + i0) * 128;
    for (int mm = 0; mm < 16; ++mm) {
        float s = pS[0][mm] + pS[1][mm] + pS[2][mm] + pS[3][mm];
        float v = part[0][mm][tid] + part[1][mm][tid] +
                  part[2][mm][tid] + part[3][mm][tid];
        orow[(size_t)mm * 128 + tid] = v / s;
    }
}

// ---------------------------------------------------------------------------
// launch
// ---------------------------------------------------------------------------
extern "C" void kernel_launch(void* const* d_in, const int* in_sizes, int n_in,
                              void* d_out, int out_size, void* d_ws, size_t ws_size,
                              hipStream_t stream) {
    const float* x    = (const float*)d_in[0];
    const int*   mask = (const int*)d_in[1];
    // d_in[2] = Wr: dead (softmax shift invariance eliminates lr_row)
    const float* Wc   = (const float*)d_in[3];
    const float* Wcat = (const float*)d_in[4];
    const float* Wx   = (const float*)d_in[5];
    const float* bx   = (const float*)d_in[6];
    float*       out  = (float*)d_out;

    char* ws = (char*)d_ws;
    float*          lrcol = (float*)ws;                        // 16384 f32  (64 KB)
    unsigned short* Ebf   = (unsigned short*)(ws + 65536);     // 16384 u16  (32 KB)
    unsigned short* swiz  = (unsigned short*)(ws + 131072);    // 2,097,152 u16 (4 MB)

    k_lrcol  <<<64,   256, 0, stream>>>(x, Wc, Wcat, lrcol);
    k_expnorm<<<8,    256, 0, stream>>>(lrcol, Ebf);
    k_xvswiz <<<512,  256, 0, stream>>>(x, Wx, bx, swiz);
    k_attn   <<<1024, 128, 0, stream>>>(mask, Ebf, swiz, out);
}
